// DualInt8LinearConv_1357209666480
// MI455X (gfx1250) — 8-bit matrix operand order repaired and checked by reading, compiled, not run on hardware
//
#include <hip/hip_runtime.h>
#include <hip/hip_bf16.h>
#include <stdint.h>

typedef __attribute__((ext_vector_type(8))) int v8i;

#define QMAXF 32767.0f

__global__ __launch_bounds__(256) void quant_rows(
    const float* __restrict__ x, unsigned char* __restrict__ alo,
    unsigned char* __restrict__ ahi, float* __restrict__ scale_x, int cin)
{
    __shared__ float red[256];
    const int row = blockIdx.x;
    const int t   = threadIdx.x;
    const float4* xr = reinterpret_cast<const float4*>(x + (size_t)row * cin);

    float4 v[4];
    float mx = 0.0f;
    #pragma unroll
    for (int j = 0; j < 4; ++j) {
        v[j] = xr[j * 256 + t];
        mx = fmaxf(mx, fmaxf(fmaxf(fabsf(v[j].x), fabsf(v[j].y)),
                             fmaxf(fabsf(v[j].z), fabsf(v[j].w))));
    }
    red[t] = mx;
    __syncthreads();
    for (int s = 128; s > 0; s >>= 1) {
        if (t < s) red[t] = fmaxf(red[t], red[t + s]);
        __syncthreads();
    }
    const float amax  = red[0];
    const float sp    = (amax > 20.0f) ? amax : log1pf(__expf(amax));
    const float scale = sp * (1.0f / QMAXF);
    if (t == 0) scale_x[row] = scale;
    const float inv = 1.0f / scale;

    uint32_t* lo32 = reinterpret_cast<uint32_t*>(alo) + (size_t)row * (cin >> 2);
    uint32_t* hi32 = reinterpret_cast<uint32_t*>(ahi) + (size_t)row * (cin >> 2);
    #pragma unroll
    for (int j = 0; j < 4; ++j) {
        const float f[4] = {v[j].x, v[j].y, v[j].z, v[j].w};
        int q[4];
        #pragma unroll
        for (int e = 0; e < 4; ++e) {
            float r = rintf(f[e] * inv);
            r = fminf(fmaxf(r, -32768.0f), 32767.0f);
            q[e] = (int)r;
        }
        const uint32_t lo =  (uint32_t)(q[0] & 0xff)        | ((uint32_t)(q[1] & 0xff) << 8)
                          | ((uint32_t)(q[2] & 0xff) << 16) | ((uint32_t)(q[3] & 0xff) << 24);
        const uint32_t hi =  (uint32_t)((q[0] >> 8) & 0xff)        | ((uint32_t)((q[1] >> 8) & 0xff) << 8)
                          | ((uint32_t)((q[2] >> 8) & 0xff) << 16) | ((uint32_t)((q[3] >> 8) & 0xff) << 24);
        lo32[j * 256 + t] = lo;
        hi32[j * 256 + t] = hi;
    }
}

__global__ __launch_bounds__(256) void pack_w(
    const float* __restrict__ w, unsigned char* __restrict__ wq, int n4)
{
    const int i = blockIdx.x * blockDim.x + threadIdx.x;
    if (i >= n4) return;
    const float4 v = reinterpret_cast<const float4*>(w)[i];
    const int a = (int)rintf(v.x), b = (int)rintf(v.y);
    const int c = (int)rintf(v.z), d = (int)rintf(v.w);
    reinterpret_cast<uint32_t*>(wq)[i] =
         (uint32_t)(a & 0xff)        | ((uint32_t)(b & 0xff) << 8)
      | ((uint32_t)(c & 0xff) << 16) | ((uint32_t)(d & 0xff) << 24);
}

__global__ __launch_bounds__(256) void dual_gemm_iu8(
    const unsigned char* __restrict__ alo, const unsigned char* __restrict__ ahi,
    const unsigned char* __restrict__ w0q, const unsigned char* __restrict__ w1q,
    const float* __restrict__ s0, const float* __restrict__ s1,
    const float* __restrict__ bias, const float* __restrict__ scale_x,
    float* __restrict__ out, int cin, int cout)
{
    __shared__ __align__(16) unsigned char lds_lo[128 * 64];
    __shared__ __align__(16) unsigned char lds_hi[128 * 64];

    const int m0   = blockIdx.x * 128;
    const int n0   = blockIdx.y * 64;
    const int t    = threadIdx.x;
    const int wave = t >> 5;
    const int lane = t & 31;
    const int wm   = (wave & 3) * 32;
    const int wn   = (wave >> 2) * 32;
    const int lm   = lane & 15;
    const int lh   = lane >> 4;

    v8i accL0[2][2], accH0[2][2], accL1[2][2], accH1[2][2];
    #pragma unroll
    for (int a = 0; a < 2; ++a)
        #pragma unroll
        for (int b = 0; b < 2; ++b)
            #pragma unroll
            for (int i = 0; i < 8; ++i) {
                accL0[a][b][i] = 0; accH0[a][b][i] = 0;
                accL1[a][b][i] = 0; accH1[a][b][i] = 0;
            }

    const int    sr   = t >> 1;
    const int    sh   = (t & 1) * 32;
    const size_t arow = (size_t)(m0 + sr) * cin;

    for (int k0 = 0; k0 < cin; k0 += 64) {
        if (k0 + 64 < cin) {
            __builtin_prefetch(w0q + (size_t)(n0 + wn + lm) * cin + k0 + 64, 0, 0);
            __builtin_prefetch(w1q + (size_t)(n0 + wn + lm) * cin + k0 + 64, 0, 0);
        }
        __syncthreads();
        {
            const uint4* pl = reinterpret_cast<const uint4*>(alo + arow + k0 + sh);
            const uint4  l0 = pl[0], l1 = pl[1];
            *reinterpret_cast<uint4*>(&lds_lo[sr * 64 + sh])      = l0;
            *reinterpret_cast<uint4*>(&lds_lo[sr * 64 + sh + 16]) = l1;
            const uint4* ph = reinterpret_cast<const uint4*>(ahi + arow + k0 + sh);
            const uint4  h0 = ph[0], h1 = ph[1];
            *reinterpret_cast<uint4*>(&lds_hi[sr * 64 + sh])      = h0;
            *reinterpret_cast<uint4*>(&lds_hi[sr * 64 + sh + 16]) = h1;
        }
        __syncthreads();

        v8i fb0[2], fb1[2];
        #pragma unroll
        for (int nt = 0; nt < 2; ++nt) {
            const size_t orow = (size_t)(n0 + wn + nt * 16 + lm) * cin + k0 + lh * 16;
            const int4 c0 = *reinterpret_cast<const int4*>(w0q + orow);
            const int4 c1 = *reinterpret_cast<const int4*>(w0q + orow + 32);
            fb0[nt][0] = c0.x; fb0[nt][1] = c0.y; fb0[nt][2] = c0.z; fb0[nt][3] = c0.w;
            fb0[nt][4] = c1.x; fb0[nt][5] = c1.y; fb0[nt][6] = c1.z; fb0[nt][7] = c1.w;
            const int4 d0 = *reinterpret_cast<const int4*>(w1q + orow);
            const int4 d1 = *reinterpret_cast<const int4*>(w1q + orow + 32);
            fb1[nt][0] = d0.x; fb1[nt][1] = d0.y; fb1[nt][2] = d0.z; fb1[nt][3] = d0.w;
            fb1[nt][4] = d1.x; fb1[nt][5] = d1.y; fb1[nt][6] = d1.z; fb1[nt][7] = d1.w;
        }

        #pragma unroll
        for (int mt = 0; mt < 2; ++mt) {
            v8i aL, aH;
            const int abase = (wm + mt * 16 + lm) * 64 + lh * 16;
            #pragma unroll
            for (int p = 0; p < 4; ++p) {
                const uint64_t dl = *reinterpret_cast<const uint64_t*>(&lds_lo[abase + (p >> 1) * 32 + (p & 1) * 8]);
                aL[2 * p]     = (int)(uint32_t)dl;
                aL[2 * p + 1] = (int)(uint32_t)(dl >> 32);
                const uint64_t dh = *reinterpret_cast<const uint64_t*>(&lds_hi[abase + (p >> 1) * 32 + (p & 1) * 8]);
                aH[2 * p]     = (int)(uint32_t)dh;
                aH[2 * p + 1] = (int)(uint32_t)(dh >> 32);
            }
            #pragma unroll
            for (int nt = 0; nt < 2; ++nt) {
                accL0[mt][nt] = __builtin_amdgcn_wmma_i32_16x16x64_iu8(
                    false, aL, true, fb0[nt], accL0[mt][nt], false, false);
                accH0[mt][nt] = __builtin_amdgcn_wmma_i32_16x16x64_iu8(
                    true,  aH, true, fb0[nt], accH0[mt][nt], false, false);
                accL1[mt][nt] = __builtin_amdgcn_wmma_i32_16x16x64_iu8(
                    false, aL, true, fb1[nt], accL1[mt][nt], false, false);
                accH1[mt][nt] = __builtin_amdgcn_wmma_i32_16x16x64_iu8(
                    true,  aH, true, fb1[nt], accH1[mt][nt], false, false);
            }
        }
    }

    #pragma unroll
    for (int mt = 0; mt < 2; ++mt) {
        #pragma unroll
        for (int nt = 0; nt < 2; ++nt) {
            const int   o   = n0 + wn + nt * 16 + lm;
            const float fs0 = s0[o], fs1 = s1[o], fb = bias[o];
            #pragma unroll
            for (int i = 0; i < 8; ++i) {
                const int   m  = m0 + wm + mt * 16 + lh * 8 + i;
                const float sx = scale_x[m];
                const float y0 = fmaf(256.0f, (float)accH0[mt][nt][i], (float)accL0[mt][nt][i]);
                const float y1 = fmaf(256.0f, (float)accH1[mt][nt][i], (float)accL1[mt][nt][i]);
                out[(size_t)m * cout + o] = (y0 * fs0 + y1 * fs1) * sx + fb;
            }
        }
    }
}

extern "C" void kernel_launch(void* const* d_in, const int* in_sizes, int n_in,
                              void* d_out, int out_size, void* d_ws, size_t ws_size,
                              hipStream_t stream)
{
    const float* x    = (const float*)d_in[0];
    const float* w0   = (const float*)d_in[1];
    const float* w1   = (const float*)d_in[2];
    const float* s0   = (const float*)d_in[3];
    const float* s1   = (const float*)d_in[4];
    const float* bias = (const float*)d_in[5];
    float* out = (float*)d_out;

    const int cout  = in_sizes[3];
    const int cin   = in_sizes[1] / cout;
    const int nrows = in_sizes[0] / cin;

    unsigned char* ws = (unsigned char*)d_ws;
    size_t off = 0;
    auto alloc = [&](size_t bytes) {
        unsigned char* p = ws + off;
        off = (off + bytes + 255) & ~(size_t)255;
        return p;
    };
    unsigned char* alo = alloc((size_t)nrows * cin);
    unsigned char* ahi = alloc((size_t)nrows * cin);
    unsigned char* w0q = alloc((size_t)cout * cin);
    unsigned char* w1q = alloc((size_t)cout * cin);
    float*         scl = (float*)alloc((size_t)nrows * sizeof(float));
    (void)ws_size; (void)n_in; (void)out_size;

    quant_rows<<<nrows, 256, 0, stream>>>(x, alo, ahi, scl, cin);

    const int wn4 = (cout * cin) / 4;
    pack_w<<<(wn4 + 255) / 256, 256, 0, stream>>>(w0, w0q, wn4);
    pack_w<<<(wn4 + 255) / 256, 256, 0, stream>>>(w1, w1q, wn4);

    dim3 grid(nrows / 128, cout / 64);
    dual_gemm_iu8<<<grid, 256, 0, stream>>>(alo, ahi, w0q, w1q,
                                            s0, s1, bias, scl, out, cin, cout);
}
